// CausalSelfAttention_22445499089699
// MI455X (gfx1250) — matrix operand order repaired and checked by reading, not compiled, not run on hardware
//
#include <hip/hip_runtime.h>

// CDNA5 / gfx1250 causal self-attention: bf16 WMMA pipeline with TDM staging.
// B=2, S=2048, D=1024, H=16, hd=64.

typedef __attribute__((ext_vector_type(16))) __bf16 v16bf;
typedef __attribute__((ext_vector_type(8)))  float  v8f;

#define BATCH   2
#define S_LEN   2048
#define DMODEL  1024
#define NHEADS  16
#define HDIM    64

union FragB { v16bf v; unsigned u[8]; unsigned short h[16]; };
union F8    { v8f v; float f[8]; };

static __device__ inline v8f zero8() {
  v8f z = {0.f, 0.f, 0.f, 0.f, 0.f, 0.f, 0.f, 0.f};
  return z;
}

// ---------------------------------------------------------------------------
// Tensor Data Mover: 2-D tile load (global -> LDS), guarded so host pass and
// toolchains without the builtin fall back to manual staging.
// D# layout per CDNA5 ISA.
// This toolchain uses the 6-arg builtin:
//   (uint32x4 g0, int32x8 g1, int32x4 g2, int32x4 g3, int32x8 gx, i32 cpol)
// ---------------------------------------------------------------------------
#if defined(__gfx1250__) && __has_builtin(__builtin_amdgcn_tensor_load_to_lds)
#define HAVE_TDM 1
typedef __attribute__((ext_vector_type(4))) unsigned tdm_u32x4;
typedef __attribute__((ext_vector_type(8))) int      tdm_i32x8;
typedef __attribute__((ext_vector_type(4))) int      tdm_i32x4;

// tile_w elements (2B each) contiguous along dim0, tile_h rows, row stride in
// elements. LDS fill is linear (no padding).
static __device__ inline void tdm_load_tile_2d(unsigned lds_off, const void* gsrc,
                                               unsigned tile_w, unsigned tile_h,
                                               unsigned long long stride_elems) {
  const unsigned long long ga = (unsigned long long)gsrc;
  tdm_u32x4 g0 = {
      1u,                                            // count=1, user mode
      lds_off,                                       // lds_addr
      (unsigned)ga,                                  // global_addr[31:0]
      (unsigned)((ga >> 32) & 0x01FFFFFFull) | 0x80000000u  // addr[56:32] | type=2
  };
  tdm_i32x8 g1 = {
      (int)(1u << 16),                 // workgroup_mask=0, data_size=1 (2 bytes)
      (int)(tile_w << 16),             // atomic_barrier=0 | tensor_dim0[15:0]
      (int)((tile_h & 0xFFFFu) << 16), // tensor_dim0[31:16]=0 | tensor_dim1[15:0]
      (int)(tile_w << 16),             // tensor_dim1[31:16]=0 | tile_dim0
      (int)tile_h,                     // tile_dim1 | tile_dim2=0
      (int)(unsigned)stride_elems,     // tensor_dim0_stride[31:0]
      (int)(unsigned)(stride_elems >> 32), // stride[47:32] | tensor_dim1_stride[15:0]=0
      0
  };
  tdm_i32x4 z4 = {0, 0, 0, 0};
  tdm_i32x8 z8 = {0, 0, 0, 0, 0, 0, 0, 0};
  __builtin_amdgcn_tensor_load_to_lds(g0, g1, z4, z4, z8, 0);
}

static __device__ inline void tdm_wait0() {
#if __has_builtin(__builtin_amdgcn_s_wait_tensorcnt)
  __builtin_amdgcn_s_wait_tensorcnt(0);
#else
  asm volatile("s_wait_tensorcnt 0x0" ::: "memory");
#endif
}
#endif

// ---------------------------------------------------------------------------
// fp32 -> bf16 conversion
// ---------------------------------------------------------------------------
__global__ __launch_bounds__(256)
void cvt_f32_to_bf16(const float* __restrict__ in, __bf16* __restrict__ out, int n) {
  int i = (blockIdx.x * 256 + threadIdx.x) * 4;
  if (i + 3 < n) {
    float4 f = *(const float4*)(in + i);
    out[i + 0] = (__bf16)f.x;
    out[i + 1] = (__bf16)f.y;
    out[i + 2] = (__bf16)f.z;
    out[i + 3] = (__bf16)f.w;
  } else {
    for (; i < n; ++i) out[i] = (__bf16)in[i];
  }
}

// ---------------------------------------------------------------------------
// bf16 GEMM:  C[M,N] = A[M,K] @ B[K,N] + bias[N]
// Block tile 128x128, K-step 32, 256 threads = 8 waves (2x4),
// each wave: 4x2 grid of 16x16 WMMA tiles via v_wmma_f32_16x16x32_bf16.
// A tile staged by TDM (wave 0); B tile staged transposed so both A and B
// fragments load as aligned ds_load_b128 pairs.
// ---------------------------------------------------------------------------
template <bool BF16_OUT>
__global__ __launch_bounds__(256)
void gemm_bias_wmma(const __bf16* __restrict__ A, const __bf16* __restrict__ B,
                    const float* __restrict__ bias, void* __restrict__ Cout,
                    int M, int N, int K) {
  __shared__ alignas(16) __bf16 lds_a[128 * 32];   // [row][k]
  __shared__ alignas(16) __bf16 lds_bt[128 * 32];  // [col][k] (transposed)

  const int tid  = threadIdx.x;
  const int lane = tid & 31;
  const int w    = tid >> 5;
  const int wr   = w >> 2;       // 0..1 -> 64 rows each
  const int wc   = w & 3;        // 0..3 -> 32 cols each
  const int lm   = lane & 15;
  const int lh   = lane >> 4;
  const int m0   = blockIdx.y * 128;
  const int n0   = blockIdx.x * 128;

  v8f acc[4][2];
#pragma unroll
  for (int mi = 0; mi < 4; ++mi)
#pragma unroll
    for (int ni = 0; ni < 2; ++ni) acc[mi][ni] = zero8();

  for (int kt = 0; kt < K; kt += 32) {
    // --- Stage A tile 128x32 ---
#ifdef HAVE_TDM
    if (w == 0) {
      tdm_load_tile_2d((unsigned)(unsigned long long)(const void*)lds_a,
                       A + (size_t)m0 * K + kt, 32u, 128u, (unsigned long long)K);
    }
#else
    {
      const int r = tid >> 1, hf = tid & 1;
      *(uint4*)&lds_a[r * 32 + hf * 16] =
          *(const uint4*)(A + (size_t)(m0 + r) * K + kt + hf * 16);
    }
#endif
    // --- Stage B tile 32x128, transposed into [col][k] ---
    {
      const int r = tid >> 3, seg = tid & 7;  // r = k row 0..31, 16 cols per thread
      union { uint4 q; __bf16 e[16]; } tb;
      tb.q = *(const uint4*)(B + (size_t)(kt + r) * N + n0 + seg * 16);
#pragma unroll
      for (int i = 0; i < 16; ++i) lds_bt[(seg * 16 + i) * 32 + r] = tb.e[i];
      if (kt + 32 < K)
        __builtin_prefetch(B + (size_t)(kt + 32 + r) * N + n0 + seg * 16, 0, 3);
    }
#ifdef HAVE_TDM
    if (w == 0) tdm_wait0();
#endif
    __syncthreads();

    // A fragments (lane m = lane%16; VGPR i -> K = (lane/16)*16 + 2i)
    FragB afr[4];
#pragma unroll
    for (int mi = 0; mi < 4; ++mi) {
      const int row = wr * 64 + mi * 16 + lm;
#pragma unroll
      for (int i = 0; i < 8; ++i) {
        const int k = (lh << 4) + (i << 1);
        afr[mi].u[i] = *(const unsigned*)&lds_a[row * 32 + k];
      }
    }
    // B fragments (lane n = lane%16; VGPR i -> K pair = (lane/16)*16 + 2i):
    // contiguous in lds_bt -> two ds_load_b128 each.
    FragB bfr[2];
#pragma unroll
    for (int ni = 0; ni < 2; ++ni) {
      const int col = wc * 32 + ni * 16 + lm;
      const unsigned* src = (const unsigned*)&lds_bt[col * 32 + lh * 16];
#pragma unroll
      for (int i = 0; i < 8; ++i) bfr[ni].u[i] = src[i];
    }
#pragma unroll
    for (int mi = 0; mi < 4; ++mi)
#pragma unroll
      for (int ni = 0; ni < 2; ++ni)
        acc[mi][ni] = __builtin_amdgcn_wmma_f32_16x16x32_bf16(
            false, afr[mi].v, false, bfr[ni].v, (short)0, acc[mi][ni], false, false);
    __syncthreads();
  }

  // Store (C layout: lane n = lane%16; VGPR v -> m = (lane/16)*8 + v)
#pragma unroll
  for (int mi = 0; mi < 4; ++mi)
#pragma unroll
    for (int ni = 0; ni < 2; ++ni) {
      const int col = n0 + wc * 32 + ni * 16 + lm;
      const float bv = bias[col];
      F8 r; r.v = acc[mi][ni];
#pragma unroll
      for (int v = 0; v < 8; ++v) {
        const int row = m0 + wr * 64 + mi * 16 + lh * 8 + v;
        const float val = r.f[v] + bv;
        if constexpr (BF16_OUT)
          ((__bf16*)Cout)[(size_t)row * N + col] = (__bf16)val;
        else
          ((float*)Cout)[(size_t)row * N + col] = val;
      }
    }
}

// ---------------------------------------------------------------------------
// Causal flash attention. qkv: [B, S, 3*D] bf16 (q | k | v).
// Grid: (S/64 q-blocks, B*H). Block: 128 threads = 4 waves; wave w owns
// query rows qb*64 + w*16 .. +15. Online softmax, 16x16x32 bf16 WMMA for
// QK^T and PV. K tile staged by TDM; V staged transposed manually.
// ---------------------------------------------------------------------------
__global__ __launch_bounds__(128)
void flash_attn_wmma(const __bf16* __restrict__ qkv, __bf16* __restrict__ outb) {
  __shared__ alignas(16) __bf16 lds_k[64 * 64];     // [key][dim]
  __shared__ alignas(16) __bf16 lds_vt[64 * 64];    // [dim][key] (transposed)
  __shared__ alignas(4)  __bf16 lds_p[4][16 * 64];  // per-wave P staging

  const int qb   = blockIdx.x;
  const int bh   = blockIdx.y;
  const int b    = bh >> 4;
  const int h    = bh & 15;
  const int tid  = threadIdx.x;
  const int w    = tid >> 5;
  const int lane = tid & 31;
  const int lm   = lane & 15;
  const int lh   = lane >> 4;
  const float scale = 0.125f;  // 1/sqrt(64)

  const size_t rstride = 3 * DMODEL;
  const __bf16* qbase = qkv + (size_t)b * S_LEN * rstride + (size_t)h * HDIM;
  const __bf16* kbase = qbase + DMODEL;
  const __bf16* vbase = qbase + 2 * DMODEL;

  // Q A-fragments for 2 K-chunks (hd = 64 = 2*32), loaded once from global.
  const int qrow_frag = qb * 64 + w * 16 + lm;
  FragB aq[2];
#pragma unroll
  for (int c = 0; c < 2; ++c)
#pragma unroll
    for (int i = 0; i < 8; ++i) {
      const int k = c * 32 + (lh << 4) + (i << 1);
      aq[c].u[i] = *(const unsigned*)(qbase + (size_t)qrow_frag * rstride + k);
    }

  float m_i[8], l_i[8];
  v8f acc_o[4];
#pragma unroll
  for (int v = 0; v < 8; ++v) { m_i[v] = -1e30f; l_i[v] = 0.f; }
#pragma unroll
  for (int dt = 0; dt < 4; ++dt) acc_o[dt] = zero8();

  // Uniform loop bound across all waves in the block (causal).
  for (int kb = 0; kb <= qb; ++kb) {
    // --- Stage K block via TDM (64 keys x 64 dims, row stride 3*D) ---
#ifdef HAVE_TDM
    if (w == 0) {
      tdm_load_tile_2d((unsigned)(unsigned long long)(const void*)lds_k,
                       kbase + (size_t)(kb * 64) * rstride, 64u, 64u,
                       (unsigned long long)rstride);
    }
#else
    {
      const int r = tid >> 1, hf = tid & 1;
      const __bf16* ksrc = kbase + (size_t)(kb * 64 + r) * rstride + hf * 32;
      *(uint4*)&lds_k[r * 64 + hf * 32]     = *(const uint4*)(ksrc);
      *(uint4*)&lds_k[r * 64 + hf * 32 + 8] = *(const uint4*)(ksrc + 8);
    }
#endif
    // --- Stage V block, transposed: lds_vt[dim][key] ---
    {
      const int r = tid >> 1, hf = tid & 1;
      union { uint4 q[2]; __bf16 e[32]; } vb;
      const __bf16* vsrc = vbase + (size_t)(kb * 64 + r) * rstride + hf * 32;
      vb.q[0] = *(const uint4*)(vsrc);
      vb.q[1] = *(const uint4*)(vsrc + 8);
#pragma unroll
      for (int i = 0; i < 32; ++i) lds_vt[(hf * 32 + i) * 64 + r] = vb.e[i];
      if (kb < qb) {
        __builtin_prefetch(vsrc + 64 * rstride, 0, 3);
        __builtin_prefetch(kbase + (size_t)((kb + 1) * 64 + r) * rstride + hf * 32, 0, 3);
      }
    }
#ifdef HAVE_TDM
    if (w == 0) tdm_wait0();
#endif
    __syncthreads();

    // S = Q @ K^T : 4 key-subtiles x 2 K-chunks
    v8f sc[4];
#pragma unroll
    for (int nt = 0; nt < 4; ++nt) sc[nt] = zero8();
#pragma unroll
    for (int nt = 0; nt < 4; ++nt)
#pragma unroll
      for (int c = 0; c < 2; ++c) {
        FragB bk;
#pragma unroll
        for (int i = 0; i < 8; ++i) {
          const int k = c * 32 + (lh << 4) + (i << 1);
          bk.u[i] = *(const unsigned*)&lds_k[(nt * 16 + lm) * 64 + k];
        }
        sc[nt] = __builtin_amdgcn_wmma_f32_16x16x32_bf16(
            false, aq[c].v, false, bk.v, (short)0, sc[nt], false, false);
      }

    // Scale + causal mask + online softmax (rows in C layout).
    float pbuf[4][8], rowmax[8];
#pragma unroll
    for (int v = 0; v < 8; ++v) rowmax[v] = -1e30f;
#pragma unroll
    for (int nt = 0; nt < 4; ++nt) {
      F8 sf; sf.v = sc[nt];
#pragma unroll
      for (int v = 0; v < 8; ++v) {
        const int q_g = qb * 64 + w * 16 + lh * 8 + v;
        const int k_g = kb * 64 + nt * 16 + lm;
        float s = sf.f[v] * scale;
        if (k_g > q_g) s = -1e30f;
        pbuf[nt][v] = s;
        rowmax[v] = fmaxf(rowmax[v], s);
      }
    }
#pragma unroll
    for (int v = 0; v < 8; ++v) {
      float r = rowmax[v];
      r = fmaxf(r, __shfl_xor(r, 1));
      r = fmaxf(r, __shfl_xor(r, 2));
      r = fmaxf(r, __shfl_xor(r, 4));
      r = fmaxf(r, __shfl_xor(r, 8));
      rowmax[v] = r;
    }
    float alpha[8], rowsum[8];
#pragma unroll
    for (int v = 0; v < 8; ++v) {
      const float nm = fmaxf(m_i[v], rowmax[v]);
      alpha[v]  = __expf(m_i[v] - nm);
      m_i[v]    = nm;
      rowsum[v] = 0.f;
    }
#pragma unroll
    for (int nt = 0; nt < 4; ++nt)
#pragma unroll
      for (int v = 0; v < 8; ++v) {
        const float p = __expf(pbuf[nt][v] - m_i[v]);
        pbuf[nt][v] = p;
        rowsum[v] += p;
      }
#pragma unroll
    for (int v = 0; v < 8; ++v) {
      float r = rowsum[v];
      r += __shfl_xor(r, 1);
      r += __shfl_xor(r, 2);
      r += __shfl_xor(r, 4);
      r += __shfl_xor(r, 8);
      l_i[v] = l_i[v] * alpha[v] + r;
    }
#pragma unroll
    for (int dt = 0; dt < 4; ++dt) {
      F8 a; a.v = acc_o[dt];
#pragma unroll
      for (int v = 0; v < 8; ++v) a.f[v] *= alpha[v];
      acc_o[dt] = a.v;
    }

    // Reshape P (C layout) -> A fragments via per-wave LDS.
#pragma unroll
    for (int nt = 0; nt < 4; ++nt)
#pragma unroll
      for (int v = 0; v < 8; ++v) {
        const int m = lh * 8 + v;
        lds_p[w][m * 64 + nt * 16 + lm] = (__bf16)pbuf[nt][v];
      }
    FragB ap[2];
#pragma unroll
    for (int c = 0; c < 2; ++c)
#pragma unroll
      for (int i = 0; i < 8; ++i) {
        const int k = c * 32 + (lh << 4) + (i << 1);
        ap[c].u[i] = *(const unsigned*)&lds_p[w][lm * 64 + k];
      }

    // O += P @ V : 4 hd-subtiles x 2 key-chunks. V fragments contiguous in lds_vt.
#pragma unroll
    for (int dt = 0; dt < 4; ++dt)
#pragma unroll
      for (int c = 0; c < 2; ++c) {
        FragB bv;
#pragma unroll
        for (int i = 0; i < 8; ++i) {
          const int n = c * 32 + (lh << 4) + (i << 1);
          bv.u[i] = *(const unsigned*)&lds_vt[(dt * 16 + lm) * 64 + n];
        }
        acc_o[dt] = __builtin_amdgcn_wmma_f32_16x16x32_bf16(
            false, ap[c].v, false, bv.v, (short)0, acc_o[dt], false, false);
      }
    __syncthreads();
  }

  // Epilogue: O / l, store bf16 to [B, S, D] layout.
#pragma unroll
  for (int dt = 0; dt < 4; ++dt) {
    F8 a; a.v = acc_o[dt];
#pragma unroll
    for (int v = 0; v < 8; ++v) {
      const int q_g = qb * 64 + w * 16 + lh * 8 + v;
      const int d   = dt * 16 + lm;
      const float val = a.f[v] / l_i[v];
      outb[(size_t)(b * S_LEN + q_g) * DMODEL + h * HDIM + d] = (__bf16)val;
    }
  }
}

// ---------------------------------------------------------------------------
// Launch
// ---------------------------------------------------------------------------
extern "C" void kernel_launch(void* const* d_in, const int* in_sizes, int n_in,
                              void* d_out, int out_size, void* d_ws, size_t ws_size,
                              hipStream_t stream) {
  const float* x      = (const float*)d_in[0];  // [2,2048,1024]
  const float* qkv_w  = (const float*)d_in[1];  // [1024,3072]
  const float* qkv_b  = (const float*)d_in[2];  // [3072]
  const float* out_w  = (const float*)d_in[3];  // [1024,1024]
  const float* out_b  = (const float*)d_in[4];  // [1024]

  const int M  = BATCH * S_LEN;        // 4096
  const int K  = DMODEL;               // 1024
  const int N1 = 3 * DMODEL;           // 3072

  // Workspace layout (bf16 segments, 256B-aligned). Total ~48 MB.
  char* ws = (char*)d_ws;
  __bf16* xb   = (__bf16*)(ws);                               // 4096*1024
  __bf16* wqkv = (__bf16*)(ws + 8388608);                     // 1024*3072
  __bf16* wo   = (__bf16*)(ws + 8388608 + 6291456);           // 1024*1024
  __bf16* qkvb = (__bf16*)(ws + 16777216);                    // 4096*3072
  __bf16* aob  = (__bf16*)(ws + 16777216 + 25165824);         // 4096*1024

  // 1) fp32 -> bf16 conversions
  {
    int n = M * K;  // 4,194,304
    cvt_f32_to_bf16<<<(n / 4 + 255) / 256, 256, 0, stream>>>(x, xb, n);
    n = K * N1;     // 3,145,728
    cvt_f32_to_bf16<<<(n / 4 + 255) / 256, 256, 0, stream>>>(qkv_w, wqkv, n);
    n = K * K;      // 1,048,576
    cvt_f32_to_bf16<<<(n / 4 + 255) / 256, 256, 0, stream>>>(out_w, wo, n);
  }

  // 2) QKV projection: [4096,1024] @ [1024,3072] + bias -> bf16
  gemm_bias_wmma<true><<<dim3(N1 / 128, M / 128), 256, 0, stream>>>(
      xb, wqkv, qkv_b, (void*)qkvb, M, N1, K);

  // 3) Causal flash attention -> bf16 [B,S,D]
  flash_attn_wmma<<<dim3(S_LEN / 64, BATCH * NHEADS), 128, 0, stream>>>(qkvb, aob);

  // 4) Output projection: [4096,1024] @ [1024,1024] + bias -> fp32 d_out
  gemm_bias_wmma<false><<<dim3(K / 128, M / 128), 256, 0, stream>>>(
      aob, wo, out_b, d_out, M, K, K);
}
